// GroupedMLP_86242943303728
// MI455X (gfx1250) — compile-verified, matrix operand order checked by reading, not run on hardware
//
#include <hip/hip_runtime.h>
#include <stdint.h>

// Problem constants (match reference)
#define T_TOKENS 16384
#define HIDDEN   2048
#define INTER    1408
#define GUPW     (2 * INTER)   // 2816
#define NEXP     8

// Tiling
#define BM 128
#define BN 64
#define BK 32

typedef __attribute__((ext_vector_type(16))) __bf16 bf16x16;
typedef __attribute__((ext_vector_type(8)))  float  f32x8;
typedef __attribute__((ext_vector_type(4)))  unsigned u32x4;
typedef __attribute__((ext_vector_type(8)))  unsigned u32x8;

union FragB16 {
    uint4   q[2];
    bf16x16 v;
};

// fp32 -> bf16 (RNE); packed hardware converter when available
__device__ __forceinline__ unsigned short f2bf(float f) {
    unsigned u = __float_as_uint(f);
    u += 0x7fffu + ((u >> 16) & 1u);
    return (unsigned short)(u >> 16);
}
__device__ __forceinline__ unsigned pk2(float lo, float hi) {
#if __has_builtin(__builtin_amdgcn_cvt_pk_bf16_f32)
    auto r = __builtin_amdgcn_cvt_pk_bf16_f32(lo, hi);
    return __builtin_bit_cast(unsigned, r);
#else
    return (unsigned)f2bf(lo) | ((unsigned)f2bf(hi) << 16);
#endif
}

// ---------------------------------------------------------------------------
// Tensor Data Mover: one instruction DMAs a 2D tile (tile_d0 x tile_d1
// elements, row pitch stride0 elements) from global to LDS. D# layout per
// CDNA5 ISA 8.3/8.4; 2-SGPR-group form (VADDR2/3 = NULL) for 2D tensors.
// Tracked by TENSORcnt. Issued by wave 0 only.
// ---------------------------------------------------------------------------
__device__ __forceinline__ void tdm_load_2d(const void* gptr, void* lptr,
                                            unsigned tensor_d0, unsigned tensor_d1,
                                            unsigned tile_d0, unsigned tile_d1,
                                            unsigned long long stride0,
                                            unsigned data_size_code) // 0=1B 1=2B 2=4B
{
    unsigned long long ga = (unsigned long long)(uintptr_t)gptr;
    u32x4 g0;
    g0[0] = 1u;                                                // count=1, user D#
    g0[1] = (unsigned)(uintptr_t)lptr;                         // lds_addr (bytes)
    g0[2] = (unsigned)(ga & 0xffffffffu);                      // global_addr[31:0]
    g0[3] = (unsigned)((ga >> 32) & 0x01ffffffu) | (2u << 30); // addr[56:32] | type=2
    u32x8 g1;
    g1[0] = data_size_code << 16;                              // wg_mask=0, data_size
    g1[1] = (tensor_d0 & 0xffffu) << 16;                       // tensor_dim0[15:0]
    g1[2] = ((tensor_d0 >> 16) & 0xffffu) | ((tensor_d1 & 0xffffu) << 16);
    g1[3] = ((tensor_d1 >> 16) & 0xffffu) | (tile_d0 << 16);   // tile_dim0
    g1[4] = tile_d1 & 0xffffu;                                 // tile_dim1, tile_dim2=0
    g1[5] = (unsigned)(stride0 & 0xffffffffu);                 // tensor_dim0_stride
    g1[6] = (unsigned)((stride0 >> 32) & 0xffffu);             // stride0 hi, stride1=0
    g1[7] = 0u;
    asm volatile("tensor_load_to_lds %0, %1" :: "s"(g0), "s"(g1) : "memory");
}
__device__ __forceinline__ void tdm_wait0() {
    __builtin_amdgcn_s_wait_tensorcnt(0);
}

// ---------------------------------------------------------------------------
// Kernel 1: h[T, INTER] = silu(X @ Wg) * (X @ Wu), h stored bf16 in ws.
// ---------------------------------------------------------------------------
__global__ __launch_bounds__(256)
void gemm1_swiglu(const float* __restrict__ x,
                  const float* __restrict__ gup,
                  const int* __restrict__ gs,
                  unsigned short* __restrict__ h)
{
    __shared__ float          As[2][BM * BK];   // fp32, filled by TDM
    __shared__ unsigned short Bg[2][BN * BK];   // [n][k] transposed bf16
    __shared__ unsigned short Bu[2][BN * BK];

    const int row0 = blockIdx.x * BM;
    const int n0   = blockIdx.y * BN;

    int e = 0, c = 0;
    for (int i = 0; i < NEXP; ++i) { c += gs[i]; if (row0 >= c) e = i + 1; }
    const float* wbase = gup + (size_t)e * HIDDEN * GUPW;
    const float* atile = x + (size_t)row0 * HIDDEN;

    const int tid  = threadIdx.x;
    const int lane = tid & 31;
    const int wv   = tid >> 5;
    const int nl   = lane & 15;
    const int kh   = (lane >> 4) * 8;

    // B mapping: thread -> one N column, 8 K values (coalesced across lanes)
    const int bn  = tid & 63;
    const int bkc = (tid >> 6) * 8;
    const float* bcol = wbase + (size_t)bkc * GUPW + n0 + bn;

    float brg[8], bru[8];

    auto issue_A = [&](int k0, int buf) {
        if (wv == 0)
            tdm_load_2d(atile + k0, &As[buf][0],
                        HIDDEN, T_TOKENS - row0, BK, BM,
                        (unsigned long long)HIDDEN, 2u /* 4B */);
    };
    auto load_B = [&](int k0) {
        const float* p = bcol + (size_t)k0 * GUPW;
#pragma unroll
        for (int j = 0; j < 8; ++j) {
            brg[j] = p[(size_t)j * GUPW];
            bru[j] = p[(size_t)j * GUPW + INTER];
        }
    };
    auto store_B = [&](int buf) {
        uint4 pg = { pk2(brg[0], brg[1]), pk2(brg[2], brg[3]),
                     pk2(brg[4], brg[5]), pk2(brg[6], brg[7]) };
        uint4 pu = { pk2(bru[0], bru[1]), pk2(bru[2], bru[3]),
                     pk2(bru[4], bru[5]), pk2(bru[6], bru[7]) };
        *(uint4*)&Bg[buf][bn * BK + bkc] = pg;
        *(uint4*)&Bu[buf][bn * BK + bkc] = pu;
    };

    f32x8 accg[4], accu[4];
#pragma unroll
    for (int j = 0; j < 4; ++j) { accg[j] = (f32x8)0.0f; accu[j] = (f32x8)0.0f; }

    issue_A(0, 0);
    load_B(0);

    const int NK = HIDDEN / BK;   // 64
    for (int ks = 0; ks < NK; ++ks) {
        const int buf = ks & 1;

        store_B(buf);                      // regs(ks) -> LDS[buf]
        if (wv == 0) tdm_wait0();          // A chunk ks landed in As[buf]
        __syncthreads();                   // retires iter ks-1 reads of buf^1

        if (ks + 1 < NK) {                 // prefetch ks+1 under WMMA burst
            issue_A((ks + 1) * BK, buf ^ 1);
            load_B((ks + 1) * BK);
        }

        // A fragment: fp32 LDS -> bf16 regs (ISA 16-bit A 16x32 layout)
        const float* ap = &As[buf][(wv * 16 + nl) * BK];
        float4 f0 = *(const float4*)(ap + kh);
        float4 f1 = *(const float4*)(ap + kh + 4);
        float4 f2 = *(const float4*)(ap + kh + 16);
        float4 f3 = *(const float4*)(ap + kh + 20);
        FragB16 a;
        a.q[0] = { pk2(f0.x, f0.y), pk2(f0.z, f0.w), pk2(f1.x, f1.y), pk2(f1.z, f1.w) };
        a.q[1] = { pk2(f2.x, f2.y), pk2(f2.z, f2.w), pk2(f3.x, f3.y), pk2(f3.z, f3.w) };

        // preload all B fragments first so DS latency retires under the WMMAs
        FragB16 bg[4], bu[4];
#pragma unroll
        for (int j = 0; j < 4; ++j) {
            bg[j].q[0] = *(const uint4*)&Bg[buf][(j * 16 + nl) * BK + kh];
            bg[j].q[1] = *(const uint4*)&Bg[buf][(j * 16 + nl) * BK + kh + 16];
            bu[j].q[0] = *(const uint4*)&Bu[buf][(j * 16 + nl) * BK + kh];
            bu[j].q[1] = *(const uint4*)&Bu[buf][(j * 16 + nl) * BK + kh + 16];
        }
#pragma unroll
        for (int j = 0; j < 4; ++j) {
            accg[j] = __builtin_amdgcn_wmma_f32_16x16x32_bf16(
                false, a.v, false, bg[j].v, (short)0, accg[j], false, false);
            accu[j] = __builtin_amdgcn_wmma_f32_16x16x32_bf16(
                false, a.v, false, bu[j].v, (short)0, accu[j], false, false);
        }
    }

    // fused SwiGLU epilogue (C/D layout: VGPR r -> M = r, hi lanes +8)
    const int mrow = row0 + wv * 16 + (lane >> 4) * 8;
#pragma unroll
    for (int j = 0; j < 4; ++j) {
        const int col = n0 + j * 16 + nl;
#pragma unroll
        for (int r = 0; r < 8; ++r) {
            float g = accg[j][r];
            float u = accu[j][r];
            float s = g * __builtin_amdgcn_rcpf(1.0f + __expf(-g));
            h[(size_t)(mrow + r) * INTER + col] = f2bf(s * u);
        }
    }
}

// ---------------------------------------------------------------------------
// Kernel 2: out[T, HIDDEN] fp32 = h(bf16) @ Wd[e] (fp32 -> bf16)
// ---------------------------------------------------------------------------
__global__ __launch_bounds__(256)
void gemm2_down(const unsigned short* __restrict__ h,
                const float* __restrict__ down,
                const int* __restrict__ gs,
                float* __restrict__ out)
{
    __shared__ unsigned short As[2][BM * BK];   // bf16, filled by TDM
    __shared__ unsigned short Bs[2][BN * BK];   // [n][k] transposed bf16

    const int row0 = blockIdx.x * BM;
    const int n0   = blockIdx.y * BN;

    int e = 0, c = 0;
    for (int i = 0; i < NEXP; ++i) { c += gs[i]; if (row0 >= c) e = i + 1; }
    const float* wbase = down + (size_t)e * INTER * HIDDEN;
    const unsigned short* atile = h + (size_t)row0 * INTER;

    const int tid  = threadIdx.x;
    const int lane = tid & 31;
    const int wv   = tid >> 5;
    const int nl   = lane & 15;
    const int kh   = (lane >> 4) * 8;

    const int bn  = tid & 63;
    const int bkc = (tid >> 6) * 8;
    const float* bcol = wbase + (size_t)bkc * HIDDEN + n0 + bn;

    float br[8];

    auto issue_A = [&](int k0, int buf) {
        if (wv == 0)
            tdm_load_2d(atile + k0, &As[buf][0],
                        INTER, T_TOKENS - row0, BK, BM,
                        (unsigned long long)INTER, 1u /* 2B */);
    };
    auto load_B = [&](int k0) {
        const float* p = bcol + (size_t)k0 * HIDDEN;
#pragma unroll
        for (int j = 0; j < 8; ++j)
            br[j] = p[(size_t)j * HIDDEN];
    };
    auto store_B = [&](int buf) {
        uint4 pb = { pk2(br[0], br[1]), pk2(br[2], br[3]),
                     pk2(br[4], br[5]), pk2(br[6], br[7]) };
        *(uint4*)&Bs[buf][bn * BK + bkc] = pb;
    };

    f32x8 acc[4];
#pragma unroll
    for (int j = 0; j < 4; ++j) acc[j] = (f32x8)0.0f;

    issue_A(0, 0);
    load_B(0);

    const int NK = INTER / BK;   // 44
    for (int ks = 0; ks < NK; ++ks) {
        const int buf = ks & 1;

        store_B(buf);
        if (wv == 0) tdm_wait0();
        __syncthreads();

        if (ks + 1 < NK) {
            issue_A((ks + 1) * BK, buf ^ 1);
            load_B((ks + 1) * BK);
        }

        FragB16 a;
        a.q[0] = *(const uint4*)&As[buf][(wv * 16 + nl) * BK + kh];
        a.q[1] = *(const uint4*)&As[buf][(wv * 16 + nl) * BK + kh + 16];

        FragB16 b[4];
#pragma unroll
        for (int j = 0; j < 4; ++j) {
            b[j].q[0] = *(const uint4*)&Bs[buf][(j * 16 + nl) * BK + kh];
            b[j].q[1] = *(const uint4*)&Bs[buf][(j * 16 + nl) * BK + kh + 16];
        }
#pragma unroll
        for (int j = 0; j < 4; ++j) {
            acc[j] = __builtin_amdgcn_wmma_f32_16x16x32_bf16(
                false, a.v, false, b[j].v, (short)0, acc[j], false, false);
        }
    }

    const int mrow = row0 + wv * 16 + (lane >> 4) * 8;
#pragma unroll
    for (int j = 0; j < 4; ++j) {
        const int col = n0 + j * 16 + nl;
#pragma unroll
        for (int r = 0; r < 8; ++r)
            out[(size_t)(mrow + r) * HIDDEN + col] = acc[j][r];
    }
}

// ---------------------------------------------------------------------------
extern "C" void kernel_launch(void* const* d_in, const int* in_sizes, int n_in,
                              void* d_out, int out_size, void* d_ws, size_t ws_size,
                              hipStream_t stream) {
    const float* x   = (const float*)d_in[0];   // [T, HIDDEN] fp32
    const float* gup = (const float*)d_in[1];   // [E, HIDDEN, 2*INTER] fp32
    const float* dwn = (const float*)d_in[2];   // [E, INTER, HIDDEN] fp32
    const int*   gs  = (const int*)d_in[3];     // [E] int32
    float* out = (float*)d_out;                 // [T, HIDDEN] fp32

    // intermediate h as bf16 in workspace: 16384*1408*2 = 46.1 MB
    unsigned short* hbuf = (unsigned short*)d_ws;

    dim3 block(256);
    dim3 g1(T_TOKENS / BM, INTER / BN);    // 128 x 22
    gemm1_swiglu<<<g1, block, 0, stream>>>(x, gup, gs, hbuf);

    dim3 g2(T_TOKENS / BM, HIDDEN / BN);   // 128 x 32
    gemm2_down<<<g2, block, 0, stream>>>(hbuf, dwn, gs, out);
}
